// SmashRNNModel_36026185679077
// MI455X (gfx1250) — compile-verified, matrix operand order checked by reading, not run on hardware
//
#include <hip/hip_runtime.h>

// ===========================================================================
// SmashRNN (hierarchical BiGRU + attention) for MI455X / gfx1250.
//
//  * All GEMMs run as bf16 WMMA (v_wmma_f32_16x16x32_bf16), f32 accumulate.
//  * B matrices are packed into fragment-native layout (32 lanes x 16 bf16
//    contiguous per (n-tile, k-step)) so the hot loop is one aligned 32B
//    load per operand + wmma. A operands are pre-converted to zero-padded
//    bf16 rows so fragments are two aligned 16B loads, no guards.
//  * GRU recurrences are launch-bound: one kernel per (level, direction)
//    iterates all T steps with the hidden state resident in LDS.
// ===========================================================================

typedef __bf16 bf16_t;
typedef __attribute__((ext_vector_type(16))) __bf16 v16bf;
typedef __attribute__((ext_vector_type(8)))  __bf16 v8bf;
typedef __attribute__((ext_vector_type(8)))  float  v8f;

__device__ __forceinline__ float sigmoidf_(float x) { return 1.f / (1.f + __expf(-x)); }

// Fragment element mapping (ISA 7.12.2, 16-bit A 16x32):
//   lane<16 -> kb=0, lane>=16 -> kb=8 ; element i -> k = kb + i + (i>=8 ? 8 : 0)
#define FRAG_ASSEMBLE(lo, hi) __builtin_shufflevector(lo, hi, 0,1,2,3,4,5,6,7,8,9,10,11,12,13,14,15)

// ---------------------------------------------------------------------------
// Pack B (f32, row-major [K][N] or two half-matrices side by side) into
// fragment-native bf16: dst[((nt*kSteps + ks)*32 + lane)*16 + i].
// Columns n < Nhalf from srcA, Nhalf <= n < 2*Nhalf from srcB (if given),
// everything else (and k >= K) zero.
// ---------------------------------------------------------------------------
__global__ void pack_b2_frag_kernel(bf16_t* dst, const float* srcA, const float* srcB,
                                    int K, int Nhalf, int kSteps, int nTiles) {
  long idx = (long)blockIdx.x * blockDim.x + threadIdx.x;
  long total = (long)nTiles * kSteps * 512;
  if (idx >= total) return;
  int i    = (int)(idx & 15);
  int lane = (int)((idx >> 4) & 31);
  long t   = idx >> 9;
  int ks   = (int)(t % kSteps);
  int nt   = (int)(t / kSteps);
  int kb = (lane >> 4) << 3;
  int k  = ks * 32 + kb + i + ((i >= 8) ? 8 : 0);
  int n  = nt * 16 + (lane & 15);
  float v = 0.f;
  if (k < K) {
    if (n < Nhalf)                 v = srcA[(long)k * Nhalf + n];
    else if (srcB && n < 2 * Nhalf) v = srcB[(long)k * Nhalf + (n - Nhalf)];
  }
  dst[idx] = (bf16_t)v;
}

// GRU recurrent weights Wh [H][3H] -> fragment layout over the padded gate
// grid: combined column c = g*Hp + j (g=gate, j<Hp), nTiles3 = 3*Hp/16.
__global__ void pack_gruwh_frag_kernel(bf16_t* dst, const float* src,
                                       int H, int Hp, int kSteps) {
  int nTiles3 = (3 * Hp) >> 4;
  long idx = (long)blockIdx.x * blockDim.x + threadIdx.x;
  long total = (long)nTiles3 * kSteps * 512;
  if (idx >= total) return;
  int i    = (int)(idx & 15);
  int lane = (int)((idx >> 4) & 31);
  long t   = idx >> 9;
  int ks   = (int)(t % kSteps);
  int nt   = (int)(t / kSteps);
  int kb = (lane >> 4) << 3;
  int k  = ks * 32 + kb + i + ((i >= 8) ? 8 : 0);
  int c  = nt * 16 + (lane & 15);
  int g  = c / Hp, j = c % Hp;
  dst[idx] = (k < H && j < H) ? (bf16_t)src[(long)k * 3 * H + g * H + j] : (bf16_t)0.f;
}

// A operand: gather (optional ids indirection) + f32->bf16 + zero pad to Kp.
__global__ void cvt_a_kernel(bf16_t* dst, const float* src, const int* ids,
                             int lda, int K, int Kp, long M) {
  long i = (long)blockIdx.x * blockDim.x + threadIdx.x;
  if (i >= M * Kp) return;
  long r = i / Kp;
  int  k = (int)(i % Kp);
  const float* row = ids ? (src + (long)ids[r] * lda) : (src + r * lda);
  dst[i] = (k < K) ? (bf16_t)row[k] : (bf16_t)0.f;
}

__global__ void concat2_kernel(float* dst, const float* a, int na, const float* b, int nb) {
  int i = blockIdx.x * blockDim.x + threadIdx.x;
  if (i < na) dst[i] = a[i];
  else if (i < na + nb) dst[i] = b[i - na];
}

__global__ void build_cat_kernel(float* cat, const float* cur, const float* prev, int B, int D) {
  int i = blockIdx.x * blockDim.x + threadIdx.x;
  if (i >= B * 3 * D) return;
  int b = i / (3 * D), c = i % (3 * D);
  float v;
  if (c < D)            v = cur[b * D + c];
  else if (c < 2 * D)   v = prev[b * D + c - D];
  else { int d = c - 2 * D; v = fabsf(cur[b * D + d] - prev[b * D + d]); }
  cat[i] = v;
}

// ---------------------------------------------------------------------------
// WMMA GEMM: C[M,N] = act(A_bf16[M,Kp] @ Bfrag + bias). One wave per 16x16
// tile; M is always a multiple of 16; Bfrag fully padded -> guard-free loop.
// ---------------------------------------------------------------------------
__global__ __launch_bounds__(32)
void gemm_wmma_kernel(const bf16_t* __restrict__ A, int Kp,
                      const bf16_t* __restrict__ Bfrag,
                      const float* __restrict__ bias,
                      float* __restrict__ C, int ldc, int N, int kSteps, int relu) {
  const int lane = threadIdx.x;
  const int mt = blockIdx.x, nt = blockIdx.y;
  const int m  = lane & 15;
  const int kb = (lane >> 4) << 3;

  const bf16_t* ap = A + (long)(mt * 16 + m) * Kp + kb;
  const bf16_t* bp = Bfrag + ((long)nt * kSteps * 32 + lane) * 16;

  v8f acc = {0.f, 0.f, 0.f, 0.f, 0.f, 0.f, 0.f, 0.f};
  for (int ks = 0; ks < kSteps; ++ks) {
    v8bf alo = *(const v8bf*)(ap);
    v8bf ahi = *(const v8bf*)(ap + 16);
    v16bf a  = FRAG_ASSEMBLE(alo, ahi);
    v16bf b  = *(const v16bf*)(bp);
    acc = __builtin_amdgcn_wmma_f32_16x16x32_bf16(false, a, false, b,
                                                  (short)0, acc, false, false);
    ap += 32;
    bp += 512;
  }
  const int n = nt * 16 + (lane & 15);
  if (n >= N) return;
  const float bv = bias ? bias[n] : 0.f;
#pragma unroll
  for (int j = 0; j < 8; ++j) {
    int r = mt * 16 + j + ((lane >> 4) << 3);
    float v = acc[j] + bv;
    if (relu) v = fmaxf(v, 0.f);
    C[(long)r * ldc + n] = v;
  }
}

// ---------------------------------------------------------------------------
// Fused-time GRU: one block owns 16 sequences, iterates all T steps.
// LDS: bf16 copy of h (WMMA A-operand, zero-padded to Kp) + f32 master.
// Per step, wave w owns gate-column tiles jt = w, w+nw, ... and computes the
// SAME tile of the r/z/n blocks (3 accumulators) so gates stay in registers.
// ---------------------------------------------------------------------------
#define GRU_MAX_KP 320   // max pad32(H)  (H3=300 -> 320)
#define GRU_MAX_HP 304   // max pad16(H)  (H3=300 -> 304)

__global__ __launch_bounds__(128)
void gru_seq_kernel(const float* __restrict__ xW, int ld_xw, int col_off,
                    const int* __restrict__ lengths,
                    const bf16_t* __restrict__ Wh,   // fragment layout, 3*nTiles tiles
                    const float* __restrict__ bh,
                    float* __restrict__ hout, int ld_hout, int hout_off,
                    int Nseq, int T, int H, int Hp, int kSteps, int reverse) {
  __shared__ __align__(16) bf16_t h_bf[16][GRU_MAX_KP];
  __shared__ float h_f[16][GRU_MAX_HP];

  const int tid  = threadIdx.x;
  const int lane = tid & 31, wid = tid >> 5, nw = blockDim.x >> 5;
  const int seqBase = blockIdx.x * 16;
  const int nTiles = Hp >> 4;
  const int m  = lane & 15;
  const int kb = (lane >> 4) << 3;
  const int rbase = (lane >> 4) << 3;

  for (int i = tid; i < 16 * GRU_MAX_KP; i += blockDim.x) (&h_bf[0][0])[i] = (bf16_t)0.f;
  for (int i = tid; i < 16 * GRU_MAX_HP; i += blockDim.x) (&h_f[0][0])[i]  = 0.f;
  __syncthreads();

  for (int step = 0; step < T; ++step) {
    const int s = reverse ? (T - 1 - step) : step;

    // ---- phase 1: hW = h@Wh (WMMA) + gate math; writes h_f / hout only ----
    for (int jt = wid; jt < nTiles; jt += nw) {
      v8f cr = {0.f,0.f,0.f,0.f,0.f,0.f,0.f,0.f};
      v8f cz = {0.f,0.f,0.f,0.f,0.f,0.f,0.f,0.f};
      v8f cn = {0.f,0.f,0.f,0.f,0.f,0.f,0.f,0.f};
      const bf16_t* bpr = Wh + ((long)(jt)              * kSteps * 32 + lane) * 16;
      const bf16_t* bpz = Wh + ((long)(nTiles + jt)     * kSteps * 32 + lane) * 16;
      const bf16_t* bpn = Wh + ((long)(2 * nTiles + jt) * kSteps * 32 + lane) * 16;
      for (int ks = 0; ks < kSteps; ++ks) {
        const bf16_t* hp = &h_bf[m][ks * 32 + kb];
        v8bf alo = *(const v8bf*)(hp);
        v8bf ahi = *(const v8bf*)(hp + 16);
        v16bf a  = FRAG_ASSEMBLE(alo, ahi);
        v16bf br = *(const v16bf*)(bpr);  bpr += 512;
        v16bf bz = *(const v16bf*)(bpz);  bpz += 512;
        v16bf bn = *(const v16bf*)(bpn);  bpn += 512;
        cr = __builtin_amdgcn_wmma_f32_16x16x32_bf16(false, a, false, br, (short)0, cr, false, false);
        cz = __builtin_amdgcn_wmma_f32_16x16x32_bf16(false, a, false, bz, (short)0, cz, false, false);
        cn = __builtin_amdgcn_wmma_f32_16x16x32_bf16(false, a, false, bn, (short)0, cn, false, false);
      }
      const int ncol = jt * 16 + (lane & 15);
      if (ncol < H) {
#pragma unroll
        for (int j = 0; j < 8; ++j) {
          int r  = j + rbase;
          int sq = seqBase + r;
          if (sq >= Nseq) continue;
          int len = lengths[sq];
          float hold = h_f[r][ncol];
          float hnew = hold;                 // frozen on padded steps
          if (s < len) {
            const float* xw = xW + ((long)sq * T + s) * ld_xw + col_off;
            float hwr = cr[j] + bh[ncol];
            float hwz = cz[j] + bh[H + ncol];
            float hwn = cn[j] + bh[2 * H + ncol];
            float rg = sigmoidf_(xw[ncol]         + hwr);
            float zg = sigmoidf_(xw[H + ncol]     + hwz);
            float ng = tanhf    (xw[2 * H + ncol] + rg * hwn);
            hnew = (1.f - zg) * ng + zg * hold;
          }
          h_f[r][ncol] = hnew;               // owner-exclusive cell
          hout[((long)sq * T + s) * ld_hout + hout_off + ncol] = hnew;
        }
      }
    }
    __syncthreads();   // all WMMA reads of h_bf complete
    // ---- phase 2: refresh bf16 copy of h for next step's A-fragments ----
    for (int jt = wid; jt < nTiles; jt += nw) {
      const int ncol = jt * 16 + (lane & 15);
#pragma unroll
      for (int j = 0; j < 8; ++j) {
        int r = j + rbase;
        h_bf[r][ncol] = (bf16_t)h_f[r][ncol];
      }
    }
    __syncthreads();   // h_bf visible before next step
  }
}

// ---------------------------------------------------------------------------
// Masked softmax attention pooling: out = sum_t softmax(tanh(h Wa + ba) v) h
// One block per sequence (T <= 32, D <= 600). 50 attention channels.
// ---------------------------------------------------------------------------
__global__ __launch_bounds__(128)
void attend_kernel(const float* __restrict__ h, int ld, const int* __restrict__ lengths,
                   const float* __restrict__ Wa, const float* __restrict__ ba,
                   const float* __restrict__ v,
                   float* __restrict__ out, int T, int D) {
  __shared__ float s_sc[64];
  __shared__ float s_a[64];
  const int seq  = blockIdx.x;
  const int lane = threadIdx.x & 31, wid = threadIdx.x >> 5, nw = blockDim.x >> 5;
  const int len  = lengths[seq];
  const float* hseq = h + (long)seq * T * ld;

  for (int t = wid; t < T; t += nw) {
    float acc = 0.f;
    for (int c = lane; c < 50; c += 32) {
      float p = ba[c];
      for (int d = 0; d < D; ++d) p += hseq[(long)t * ld + d] * Wa[d * 50 + c];
      acc += v[c] * tanhf(p);
    }
    for (int off = 16; off; off >>= 1) acc += __shfl_xor(acc, off, 32);
    if (lane == 0) s_sc[t] = (t < len) ? acc : -1e9f;
  }
  __syncthreads();

  if (wid == 0) {
    float sc = (lane < T) ? s_sc[lane] : -1e9f;
    float mx = sc;
    for (int off = 16; off; off >>= 1) mx = fmaxf(mx, __shfl_xor(mx, off, 32));
    float e = (lane < T) ? __expf(sc - mx) : 0.f;
    float sm = e;
    for (int off = 16; off; off >>= 1) sm += __shfl_xor(sm, off, 32);
    if (lane < T) s_a[lane] = (lane < len) ? (e / sm) : 0.f;
  }
  __syncthreads();

  for (int d = threadIdx.x; d < D; d += blockDim.x) {
    float acc = 0.f;
    for (int t = 0; t < T; ++t) acc += s_a[t] * hseq[(long)t * ld + d];
    out[(long)seq * D + d] = acc;
  }
}

// Final head: sigmoid(hmlp @ cW2 + cB2), one wave per row.
__global__ __launch_bounds__(32)
void final_kernel(const float* __restrict__ hmlp, const float* __restrict__ w2,
                  const float* __restrict__ b2, float* __restrict__ out, int K) {
  const int row = blockIdx.x, lane = threadIdx.x;
  float acc = 0.f;
  for (int k = lane; k < K; k += 32) acc += hmlp[(long)row * K + k] * w2[k];
  for (int off = 16; off; off >>= 1) acc += __shfl_xor(acc, off, 32);
  if (lane == 0) out[row] = sigmoidf_(acc + b2[0]);
}

// ===========================================================================
// Host orchestration
// ===========================================================================
extern "C" void kernel_launch(void* const* d_in, const int* in_sizes, int n_in,
                              void* d_out, int out_size, void* d_ws, size_t ws_size,
                              hipStream_t stream) {
  (void)in_sizes; (void)n_in; (void)out_size; (void)ws_size;

  // ---- inputs (setup_inputs order) ----
  const float* emb  = (const float*)d_in[0];
  const float* wWiF = (const float*)d_in[1];  const float* wWhF = (const float*)d_in[2];
  const float* wbiF = (const float*)d_in[3];  const float* wbhF = (const float*)d_in[4];
  const float* wWiB = (const float*)d_in[5];  const float* wWhB = (const float*)d_in[6];
  const float* wbiB = (const float*)d_in[7];  const float* wbhB = (const float*)d_in[8];
  const float* sWiF = (const float*)d_in[9];  const float* sWhF = (const float*)d_in[10];
  const float* sbiF = (const float*)d_in[11]; const float* sbhF = (const float*)d_in[12];
  const float* sWiB = (const float*)d_in[13]; const float* sWhB = (const float*)d_in[14];
  const float* sbiB = (const float*)d_in[15]; const float* sbhB = (const float*)d_in[16];
  const float* pWiF = (const float*)d_in[17]; const float* pWhF = (const float*)d_in[18];
  const float* pbiF = (const float*)d_in[19]; const float* pbhF = (const float*)d_in[20];
  const float* pWiB = (const float*)d_in[21]; const float* pWhB = (const float*)d_in[22];
  const float* pbiB = (const float*)d_in[23]; const float* pbhB = (const float*)d_in[24];
  const float* waW = (const float*)d_in[25]; const float* wab = (const float*)d_in[26]; const float* wav = (const float*)d_in[27];
  const float* saW = (const float*)d_in[28]; const float* sab = (const float*)d_in[29]; const float* sav = (const float*)d_in[30];
  const float* paW = (const float*)d_in[31]; const float* pab = (const float*)d_in[32]; const float* pav = (const float*)d_in[33];
  const float* cW1 = (const float*)d_in[34]; const float* cB1 = (const float*)d_in[35];
  const float* cW2 = (const float*)d_in[36]; const float* cB2 = (const float*)d_in[37];
  // d_in[38] = click_rate (unused by reference)
  const int* ids_cur  = (const int*)d_in[39]; const int* wps_cur  = (const int*)d_in[40];
  const int* spp_cur  = (const int*)d_in[41]; const int* ppd_cur  = (const int*)d_in[42];
  const int* ids_prev = (const int*)d_in[43]; const int* wps_prev = (const int*)d_in[44];
  const int* spp_prev = (const int*)d_in[45]; const int* ppd_prev = (const int*)d_in[46];

  // ---- shapes ----
  const int B = 16, P = 8, S = 12, W = 32, E = 300;
  const int H1 = 100, H2 = 200, H3 = 300;
  const int NW = B * P * S;   // 1536 word-level sequences
  const int NT = NW * W;      // 49152 tokens
  const int NS = B * P;       // 128 sentence-level sequences
  const int H1p = 112, H2p = 208, H3p = 304;   // pad16(H)

  // ---- workspace carve-up ----
  char* ws = (char*)d_ws;
  size_t off = 0;
  auto alloc = [&](size_t bytes) -> void* {
    void* p = (void*)(ws + off);
    off += (bytes + 255) & ~(size_t)255;
    return p;
  };
  // fragment-packed bf16 weights
  bf16_t* wWi   = (bf16_t*)alloc(38L  * 10 * 512 * 2);  // [WiF|WiB], N=600, Kp=320
  bf16_t* wWhFp = (bf16_t*)alloc(21L  * 4  * 512 * 2);  // 3*112/16 tiles, Kp=128
  bf16_t* wWhBp = (bf16_t*)alloc(21L  * 4  * 512 * 2);
  bf16_t* sWi   = (bf16_t*)alloc(75L  * 7  * 512 * 2);  // N=1200, Kp=224
  bf16_t* sWhFp = (bf16_t*)alloc(39L  * 7  * 512 * 2);  // 3*208/16, Kp=224
  bf16_t* sWhBp = (bf16_t*)alloc(39L  * 7  * 512 * 2);
  bf16_t* pWi   = (bf16_t*)alloc(113L * 13 * 512 * 2);  // N=1800, Kp=416
  bf16_t* pWhFp = (bf16_t*)alloc(57L  * 10 * 512 * 2);  // 3*304/16, Kp=320
  bf16_t* pWhBp = (bf16_t*)alloc(57L  * 10 * 512 * 2);
  bf16_t* cW1p  = (bf16_t*)alloc(57L  * 57 * 512 * 2);  // N=900, Kp=1824
  // combined biases
  float* bw = (float*)alloc(600 * 4);
  float* bs = (float*)alloc(1200 * 4);
  float* bp = (float*)alloc(1800 * 4);
  // bf16 A operands
  bf16_t* tok_bf  = (bf16_t*)alloc((long)NT * 320 * 2);
  bf16_t* sent_bf = (bf16_t*)alloc((long)NW * 224 * 2);
  bf16_t* para_bf = (bf16_t*)alloc((long)NS * 416 * 2);
  bf16_t* cat_bf  = (bf16_t*)alloc(16L * 1824 * 2);
  // f32 activations (reused sequentially for cur / prev)
  float* xw_w = (float*)alloc((long)NT * 600 * 4);
  float* h_w  = (float*)alloc((long)NT * 200 * 4);
  float* sent = (float*)alloc((long)NW * 200 * 4);
  float* xw_s = (float*)alloc((long)NW * 1200 * 4);
  float* h_s  = (float*)alloc((long)NW * 400 * 4);
  float* para = (float*)alloc((long)NS * 400 * 4);
  float* xw_p = (float*)alloc((long)NS * 1800 * 4);
  float* h_p  = (float*)alloc((long)NS * 600 * 4);
  float* doc_cur  = (float*)alloc(16L * 600 * 4);
  float* doc_prev = (float*)alloc(16L * 600 * 4);
  float* cat  = (float*)alloc(16L * 1800 * 4);
  float* hmlp = (float*)alloc(16L * 900 * 4);

  auto g1d = [](long n) { return dim3((unsigned)((n + 255) / 256)); };

  // ---- pack weights into fragment layout (deterministic each call) ----
  pack_b2_frag_kernel<<<g1d(38L * 10 * 512), 256, 0, stream>>>(wWi, wWiF, wWiB, 300, 300, 10, 38);
  pack_b2_frag_kernel<<<g1d(75L * 7 * 512), 256, 0, stream>>>(sWi, sWiF, sWiB, 200, 600, 7, 75);
  pack_b2_frag_kernel<<<g1d(113L * 13 * 512), 256, 0, stream>>>(pWi, pWiF, pWiB, 400, 900, 13, 113);
  pack_b2_frag_kernel<<<g1d(57L * 57 * 512), 256, 0, stream>>>(cW1p, cW1, nullptr, 1800, 900, 57, 57);
  pack_gruwh_frag_kernel<<<g1d(21L * 4 * 512), 256, 0, stream>>>(wWhFp, wWhF, H1, H1p, 4);
  pack_gruwh_frag_kernel<<<g1d(21L * 4 * 512), 256, 0, stream>>>(wWhBp, wWhB, H1, H1p, 4);
  pack_gruwh_frag_kernel<<<g1d(39L * 7 * 512), 256, 0, stream>>>(sWhFp, sWhF, H2, H2p, 7);
  pack_gruwh_frag_kernel<<<g1d(39L * 7 * 512), 256, 0, stream>>>(sWhBp, sWhB, H2, H2p, 7);
  pack_gruwh_frag_kernel<<<g1d(57L * 10 * 512), 256, 0, stream>>>(pWhFp, pWhF, H3, H3p, 10);
  pack_gruwh_frag_kernel<<<g1d(57L * 10 * 512), 256, 0, stream>>>(pWhBp, pWhB, H3, H3p, 10);
  concat2_kernel<<<g1d(600), 256, 0, stream>>>(bw, wbiF, 300, wbiB, 300);
  concat2_kernel<<<g1d(1200), 256, 0, stream>>>(bs, sbiF, 600, sbiB, 600);
  concat2_kernel<<<g1d(1800), 256, 0, stream>>>(bp, pbiF, 900, pbiB, 900);

  // ---- per-document pipeline ----
  auto run_doc = [&](const int* ids, const int* wps, const int* spp,
                     const int* ppd, float* doc) {
    // 1) embedding gather -> bf16 tokens, then word input projection [49152,600]
    cvt_a_kernel<<<g1d((long)NT * 320), 256, 0, stream>>>(tok_bf, emb, ids, E, 300, 320, NT);
    gemm_wmma_kernel<<<dim3(NT / 16, 38), 32, 0, stream>>>(
        tok_bf, 320, wWi, bw, xw_w, 600, 600, 10, 0);
    // 2) word BiGRU (T=32, H=100)
    gru_seq_kernel<<<dim3((NW + 15) / 16), 128, 0, stream>>>(
        xw_w, 600, 0,   wps, wWhFp, wbhF, h_w, 200, 0,   NW, W, H1, H1p, 4, 0);
    gru_seq_kernel<<<dim3((NW + 15) / 16), 128, 0, stream>>>(
        xw_w, 600, 300, wps, wWhBp, wbhB, h_w, 200, 100, NW, W, H1, H1p, 4, 1);
    // 3) word attention -> sentence vectors [1536,200]
    attend_kernel<<<dim3(NW), 128, 0, stream>>>(h_w, 200, wps, waW, wab, wav, sent, W, 200);
    // 4) sentence input projection [1536,1200]
    cvt_a_kernel<<<g1d((long)NW * 224), 256, 0, stream>>>(sent_bf, sent, nullptr, 200, 200, 224, NW);
    gemm_wmma_kernel<<<dim3(NW / 16, 75), 32, 0, stream>>>(
        sent_bf, 224, sWi, bs, xw_s, 1200, 1200, 7, 0);
    // 5) sentence BiGRU (T=12, H=200)
    gru_seq_kernel<<<dim3((NS + 15) / 16), 128, 0, stream>>>(
        xw_s, 1200, 0,   spp, sWhFp, sbhF, h_s, 400, 0,   NS, S, H2, H2p, 7, 0);
    gru_seq_kernel<<<dim3((NS + 15) / 16), 128, 0, stream>>>(
        xw_s, 1200, 600, spp, sWhBp, sbhB, h_s, 400, 200, NS, S, H2, H2p, 7, 1);
    // 6) sentence attention -> paragraph vectors [128,400]
    attend_kernel<<<dim3(NS), 128, 0, stream>>>(h_s, 400, spp, saW, sab, sav, para, S, 400);
    // 7) paragraph input projection [128,1800]
    cvt_a_kernel<<<g1d((long)NS * 416), 256, 0, stream>>>(para_bf, para, nullptr, 400, 400, 416, NS);
    gemm_wmma_kernel<<<dim3(NS / 16, 113), 32, 0, stream>>>(
        para_bf, 416, pWi, bp, xw_p, 1800, 1800, 13, 0);
    // 8) paragraph BiGRU (T=8, H=300)
    gru_seq_kernel<<<dim3(1), 128, 0, stream>>>(
        xw_p, 1800, 0,   ppd, pWhFp, pbhF, h_p, 600, 0,   B, P, H3, H3p, 10, 0);
    gru_seq_kernel<<<dim3(1), 128, 0, stream>>>(
        xw_p, 1800, 900, ppd, pWhBp, pbhB, h_p, 600, 300, B, P, H3, H3p, 10, 1);
    // 9) paragraph attention -> document vector [16,600]
    attend_kernel<<<dim3(B), 128, 0, stream>>>(h_p, 600, ppd, paW, pab, pav, doc, P, 600);
  };

  run_doc(ids_cur,  wps_cur,  spp_cur,  ppd_cur,  doc_cur);
  run_doc(ids_prev, wps_prev, spp_prev, ppd_prev, doc_prev);

  // ---- classifier head ----
  build_cat_kernel<<<g1d(16L * 1800), 256, 0, stream>>>(cat, doc_cur, doc_prev, B, 600);
  cvt_a_kernel<<<g1d(16L * 1824), 256, 0, stream>>>(cat_bf, cat, nullptr, 1800, 1800, 1824, 16);
  gemm_wmma_kernel<<<dim3(1, 57), 32, 0, stream>>>(
      cat_bf, 1824, cW1p, cB1, hmlp, 900, 900, 57, 1);
  final_kernel<<<dim3(B), 32, 0, stream>>>(hmlp, cW2, cB2, (float*)d_out, 900);
}
